// Decoder_57002805952963
// MI455X (gfx1250) — matrix operand order repaired and checked by reading, not compiled, not run on hardware
//
#include <hip/hip_runtime.h>
#include <cmath>

// Chunked Gated Delta Rule linear attention for gfx1250 (MI455X).
// All matmuls via v_wmma_f32_16x16x32_bf16; fp32 running state with bf16
// transposed shadow so every WMMA operand fragment is contiguous in LDS.

typedef __bf16 bf16_t;
typedef __attribute__((ext_vector_type(16))) __bf16 v16bf;
typedef __attribute__((ext_vector_type(8)))  float  v8f;

namespace {
constexpr int B_ = 4, L_ = 4096, H_ = 16, DK = 128, DV = 128;
constexpr int C_ = 32;              // chunk length
constexpr int NC = L_ / C_;         // 128 chunks
constexpr int NT = 256;             // 8 wave32 per workgroup
constexpr int LQ  = DK + 8;         // bf16 row stride: sQ/sK   (time-major)
constexpr int LT  = C_ + 8;         // bf16 row stride: sDt/sKt (transposed)
constexpr int LST = DK + 8;         // bf16 row stride: sSt     (S^T shadow)
constexpr int LV  = DV + 4;         // f32 row stride:  sV/sS
constexpr int LAS = 18;             // f32 stride: sA (diag 16x16 blocks only)
constexpr int LAB = C_ + 8;         // bf16 stride: sAb / P
}

// ---- WMMA fragment loaders, per CDNA5 ISA VGPR layouts (CDNA5 ISA) ----

// A-matrix 16x32 bf16, row-major source: contiguous 2x16B per lane.
__device__ __forceinline__ v16bf load_a_bf(const bf16_t* s, int ld, int r0, int k0, int lane) {
  const int m  = lane & 15;
  const int kb = (lane >> 4) << 3;          // 0 or 8
  const bf16_t* p = s + (r0 + m) * ld + k0 + kb;
  v16bf a;
#pragma unroll
  for (int e = 0; e < 8; ++e) a[e] = p[e];          // K = kb + e
#pragma unroll
  for (int e = 0; e < 8; ++e) a[8 + e] = p[16 + e]; // K = 16 + kb + e
  return a;
}

// B-matrix 32x16 bf16 from a TRANSPOSED source Xt[N][K]: B[k][n] = Xt[c0+n][k0+k].
// Contiguous 2x16B per lane.
__device__ __forceinline__ v16bf load_bt_bf(const bf16_t* s, int ld, int k0, int c0, int lane) {
  const int n  = lane & 15;
  const int kb = (lane >> 4) << 3;          // 0 or 8
  const bf16_t* p = s + (c0 + n) * ld + k0 + kb;
  v16bf b;
#pragma unroll
  for (int e = 0; e < 8; ++e) b[e] = p[e];          // K = kb + e
#pragma unroll
  for (int e = 0; e < 8; ++e) b[8 + e] = p[16 + e]; // K = 16 + kb + e
  return b;
}

__global__ __launch_bounds__(NT, 1)
void gdr_chunk_kernel(const float* __restrict__ gq, const float* __restrict__ gk,
                      const float* __restrict__ gv, const float* __restrict__ gg,
                      const float* __restrict__ gb, float* __restrict__ gout) {
  __shared__ bf16_t sQ[C_ * LQ];      // normalized q (time-major, bf16)
  __shared__ bf16_t sK[C_ * LQ];      // normalized k (time-major, bf16)
  __shared__ bf16_t sDt[DV * LT];     // pseudo-values D, TRANSPOSED [n][t]
  __shared__ bf16_t sKt[DK * LT];     // (e^{a_last-a_s} * k_s)[m], TRANSPOSED [m][s]
  __shared__ bf16_t sSt[DV * LST];    // bf16 shadow of S^T: [n][m]
  __shared__ float  sV[C_ * LV];      // v -> B rows -> D rows (fp32, in place)
  __shared__ float  sS[DK * LV];      // running state S, fp32
  __shared__ float  sA[C_ * LAS];     // solve matrix, diagonal 16x16 blocks only
  __shared__ bf16_t sAb[C_ * LAB];    // bf16 A (diag blocks zeroed); reused as P
  __shared__ float  s_a[C_], s_ea[C_], s_b[C_], s_w[C_];

  const int tid  = threadIdx.x;
  const int lane = tid & 31;
  const int wid  = tid >> 5;
  const int bh   = blockIdx.x;
  const int b    = bh / H_;
  const int h    = bh % H_;

  for (int i = tid; i < DK * LV; i += NT) sS[i] = 0.f;
  for (int i = tid; i < DV * LST; i += NT) sSt[i] = (bf16_t)0.f;
  for (int i = tid; i < DV * LT; i += NT) sDt[i] = (bf16_t)0.f;
  __syncthreads();

  const int r  = tid >> 3;          // chunk row 0..31 (8 threads/row)
  const int su = tid & 7;           // 16 floats per thread
  const float inv_sqrt_dk = 0.08838834764831845f;  // 128^-0.5

  for (int ic = 0; ic < NC; ++ic) {
    const int t0 = ic * C_;

    // ---- stage chunk: load, l2norm(q)*K^-0.5, l2norm(k), v, g, beta ----
    {
      const long rb = ((long)b * L_ + t0 + r) * H_ + h;
      const float* qp = gq + rb * DK + su * 16;
      const float* kp = gk + rb * DK + su * 16;
      const float* vp = gv + rb * DK + su * 16;
      float xq[16], xk[16];
      float ssq = 0.f, ssk = 0.f;
#pragma unroll
      for (int i = 0; i < 4; ++i) {
        float4 t4 = ((const float4*)qp)[i];
        xq[4*i+0] = t4.x; xq[4*i+1] = t4.y; xq[4*i+2] = t4.z; xq[4*i+3] = t4.w;
        float4 u4 = ((const float4*)kp)[i];
        xk[4*i+0] = u4.x; xk[4*i+1] = u4.y; xk[4*i+2] = u4.z; xk[4*i+3] = u4.w;
      }
#pragma unroll
      for (int i = 0; i < 16; ++i) { ssq += xq[i]*xq[i]; ssk += xk[i]*xk[i]; }
#pragma unroll
      for (int m = 1; m < 8; m <<= 1) { ssq += __shfl_xor(ssq, m); ssk += __shfl_xor(ssk, m); }
      const float nq = rsqrtf(ssq + 1e-6f) * inv_sqrt_dk;
      const float nk = rsqrtf(ssk + 1e-6f);
#pragma unroll
      for (int i = 0; i < 16; ++i) {
        sQ[r*LQ + su*16 + i] = (bf16_t)(xq[i]*nq);
        sK[r*LQ + su*16 + i] = (bf16_t)(xk[i]*nk);
      }
#pragma unroll
      for (int i = 0; i < 4; ++i) {
        float4 t4 = ((const float4*)vp)[i];
        sV[r*LV + su*16 + 4*i+0] = t4.x; sV[r*LV + su*16 + 4*i+1] = t4.y;
        sV[r*LV + su*16 + 4*i+2] = t4.z; sV[r*LV + su*16 + 4*i+3] = t4.w;
      }
      if (tid < C_) {
        const long gi = ((long)b * L_ + t0 + tid) * H_ + h;
        s_a[tid] = gg[gi];
        s_b[tid] = gb[gi];
      }
      if (ic + 1 < NC) {  // global_prefetch_b8 for next chunk
        const long nb = (((long)b * L_ + t0 + C_ + r) * H_ + h) * DK + su * 16;
        __builtin_prefetch(gq + nb, 0, 3);
        __builtin_prefetch(gk + nb, 0, 3);
        __builtin_prefetch(gv + nb, 0, 3);
      }
    }
    __syncthreads();
    if (tid == 0) {  // inclusive cumsum of log-decay; e^a and tail weights
      float run = 0.f;
      for (int t = 0; t < C_; ++t) { run += s_a[t]; s_a[t] = run; s_ea[t] = __expf(run); }
      for (int t = 0; t < C_; ++t) s_w[t] = __expf(run - s_a[t]);
    }
    __syncthreads();

    // ---- transpose + scale K for the state update: sKt[m][s] = w_s * k_s[m] ----
    {
      const int m   = tid >> 1;          // 0..127
      const int k0t = (tid & 1) * 16;    // 0 or 16
#pragma unroll
      for (int e = 0; e < 16; ++e) {
        const int kk = k0t + e;
        sKt[m*LT + kk] = (bf16_t)((float)sK[kk*LQ + m] * s_w[kk]);
      }
    }

    // ---- Tk = K @ S; B = beta*(V - e^a * Tk) in place into sV ----
    // ---- Tq = e^a * (Q @ S) kept in registers (same tile map as output) ----
    v8f tq[2];
#pragma unroll
    for (int it = 0; it < 2; ++it) {
      const int tm = it, tn = wid;
      v8f acck = {}, accq = {};
#pragma unroll
      for (int k0 = 0; k0 < DK; k0 += 32) {
        v16bf bb = load_bt_bf(sSt, LST, k0, tn*16, lane);
        v16bf ak = load_a_bf(sK, LQ, tm*16, k0, lane);
        acck = __builtin_amdgcn_wmma_f32_16x16x32_bf16(false, ak, false, bb, (short)0, acck, false, false);
        v16bf aq = load_a_bf(sQ, LQ, tm*16, k0, lane);
        accq = __builtin_amdgcn_wmma_f32_16x16x32_bf16(false, aq, false, bb, (short)0, accq, false, false);
      }
      const int n  = tn*16 + (lane & 15);
      const int mh = (lane >> 4) << 3;
#pragma unroll
      for (int j = 0; j < 8; ++j) {
        const int t = tm*16 + mh + j;
        sV[t*LV + n] = s_b[t] * (sV[t*LV + n] - s_ea[t] * acck[j]);
        accq[j] *= s_ea[t];
      }
      tq[it] = accq;
    }
    __syncthreads();

    // ---- A[t,s] = beta_t * e^(a_t - a_s) * (k_t . k_s), strictly lower ----
    if (wid < 4) {
      const int tm = wid >> 1, tn = wid & 1;
      v8f acc = {};
#pragma unroll
      for (int k0 = 0; k0 < DK; k0 += 32) {
        v16bf a  = load_a_bf(sK, LQ, tm*16, k0, lane);
        v16bf bb = load_bt_bf(sK, LQ, k0, tn*16, lane);   // B = K^T (row-major K is its own transpose source)
        acc = __builtin_amdgcn_wmma_f32_16x16x32_bf16(false, a, false, bb, (short)0, acc, false, false);
      }
      const int s  = tn*16 + (lane & 15);
      const int mh = (lane >> 4) << 3;
#pragma unroll
      for (int j = 0; j < 8; ++j) {
        const int t = tm*16 + mh + j;
        const float av = (s < t) ? s_b[t] * __expf(s_a[t] - s_a[s]) * acc[j] : 0.f;
        const bool diag = (s >> 4) == (t >> 4);
        if (diag) sA[t*LAS + (s & 15)] = av;              // only diag blocks needed by substitution
        sAb[t*LAB + s] = diag ? (bf16_t)0.f : (bf16_t)av; // off-diag blocks for WMMA correction
      }
    }
    __syncthreads();

    // ---- solve (I + A) D = B : WMMA block correction + 16-row substitution ----
    for (int i = 0; i < C_/16; ++i) {
      if (i > 0) {
        v8f acc = {};
        v16bf a  = load_a_bf(sAb, LAB, i*16, 0, lane);    // K = 32 (earlier blocks only)
        v16bf bb = load_bt_bf(sDt, LT, 0, wid*16, lane);
        acc = __builtin_amdgcn_wmma_f32_16x16x32_bf16(false, a, false, bb, (short)0, acc, false, false);
        const int n  = wid*16 + (lane & 15);
        const int mh = (lane >> 4) << 3;
#pragma unroll
        for (int j = 0; j < 8; ++j) sV[(i*16 + mh + j)*LV + n] -= acc[j];
        __syncthreads();
      }
      if (tid < DV) {  // per-column independent forward substitution
        const int col = tid;
        for (int t = i*16; t < i*16 + 16; ++t) {
          float val = sV[t*LV + col];
          for (int s = i*16; s < t; ++s) val -= sA[t*LAS + (s & 15)] * sV[s*LV + col];
          sV[t*LV + col] = val;
          sDt[col*LT + t] = (bf16_t)val;                  // transposed store
        }
      }
      __syncthreads();
    }

    // ---- P[t,s] = e^(a_t - a_s) * (q_t . k_s), s <= t (reuse sAb) ----
    if (wid < 4) {
      const int tm = wid >> 1, tn = wid & 1;
      v8f acc = {};
#pragma unroll
      for (int k0 = 0; k0 < DK; k0 += 32) {
        v16bf a  = load_a_bf(sQ, LQ, tm*16, k0, lane);
        v16bf bb = load_bt_bf(sK, LQ, k0, tn*16, lane);
        acc = __builtin_amdgcn_wmma_f32_16x16x32_bf16(false, a, false, bb, (short)0, acc, false, false);
      }
      const int s  = tn*16 + (lane & 15);
      const int mh = (lane >> 4) << 3;
#pragma unroll
      for (int j = 0; j < 8; ++j) {
        const int t = tm*16 + mh + j;
        const float pv = (s <= t) ? __expf(s_a[t] - s_a[s]) * acc[j] : 0.f;
        sAb[t*LAB + s] = (bf16_t)pv;
      }
    }
    __syncthreads();

    // ---- O = Tq + P @ D -> global ----
#pragma unroll
    for (int it = 0; it < 2; ++it) {
      const int tm = it, tn = wid;
      v8f acc = {};
      v16bf a  = load_a_bf(sAb, LAB, tm*16, 0, lane);
      v16bf bb = load_bt_bf(sDt, LT, 0, tn*16, lane);
      acc = __builtin_amdgcn_wmma_f32_16x16x32_bf16(false, a, false, bb, (short)0, acc, false, false);
      const int n  = tn*16 + (lane & 15);
      const int mh = (lane >> 4) << 3;
#pragma unroll
      for (int j = 0; j < 8; ++j) {
        const int t = tm*16 + mh + j;
        gout[(((long)b * L_ + t0 + t) * H_ + h) * DV + n] = tq[it][j] + acc[j];
      }
    }

    // ---- S = e^{a_last} * S + (w .* K)^T @ D ; refresh bf16 shadow S^T ----
    const float elast = s_ea[C_ - 1];
    for (int tile = wid; tile < 64; tile += 8) {
      const int tm = tile >> 3, tn = tile & 7;
      v8f acc = {};
      v16bf a  = load_a_bf(sKt, LT, tm*16, 0, lane);      // contiguous, pre-scaled
      v16bf bb = load_bt_bf(sDt, LT, 0, tn*16, lane);
      acc = __builtin_amdgcn_wmma_f32_16x16x32_bf16(false, a, false, bb, (short)0, acc, false, false);
      const int n  = tn*16 + (lane & 15);
      const int mh = (lane >> 4) << 3;
#pragma unroll
      for (int j = 0; j < 8; ++j) {
        const int mm = tm*16 + mh + j;
        const float nv = elast * sS[mm*LV + n] + acc[j];
        sS[mm*LV + n] = nv;
        sSt[n*LST + mm] = (bf16_t)nv;                     // 8 contiguous bf16 -> one b128
      }
    }
    __syncthreads();
  }
}

extern "C" void kernel_launch(void* const* d_in, const int* in_sizes, int n_in,
                              void* d_out, int out_size, void* d_ws, size_t ws_size,
                              hipStream_t stream) {
  (void)in_sizes; (void)n_in; (void)out_size; (void)d_ws; (void)ws_size;
  const float* q    = (const float*)d_in[0];
  const float* k    = (const float*)d_in[1];
  const float* v    = (const float*)d_in[2];
  const float* g    = (const float*)d_in[3];
  const float* beta = (const float*)d_in[4];
  float* out = (float*)d_out;
  dim3 grid(B_ * H_);
  dim3 block(NT);
  gdr_chunk_kernel<<<grid, block, 0, stream>>>(q, k, v, g, beta, out);
}
